// MixtureOfExperts_42855183680108
// MI455X (gfx1250) — matrix operand order repaired and checked by reading, not compiled, not run on hardware
//
#include <hip/hip_runtime.h>

// ---- problem constants (match reference) ----
#define B_TOK 16384
#define DDIM  3072
#define HDIM  1024
#define NEXP  8
#define TOPK  2

// ---- WMMA / vector types ----
typedef __attribute__((ext_vector_type(16))) __bf16          bf16x16;
typedef __attribute__((ext_vector_type(8)))  float           floatx8;
typedef __attribute__((ext_vector_type(8)))  unsigned short  ushortx8;

union Frag16 {
  bf16x16  v;
  ushortx8 h[2];
};

// float -> bf16 bits, round-to-nearest-even
__device__ __forceinline__ unsigned short f2bf(float f) {
  unsigned int u = __float_as_uint(f);
  unsigned int r = u + 0x7FFFu + ((u >> 16) & 1u);
  return (unsigned short)(r >> 16);
}

// two floats -> packed bf16x2 in one 32-bit word (RNE), no 16-bit reg ops
__device__ __forceinline__ unsigned pack2bf(float a, float b) {
  unsigned ua = __float_as_uint(a);
  ua += 0x7FFFu + ((ua >> 16) & 1u);
  unsigned ub = __float_as_uint(b);
  ub += 0x7FFFu + ((ub >> 16) & 1u);
  return (ua >> 16) | (ub & 0xFFFF0000u);
}

// =====================================================================
// Kernel 1: gating. One wave32 per token.
// =====================================================================
__global__ __launch_bounds__(256) void moe_gate_kernel(
    const float* __restrict__ x, const float* __restrict__ gw,
    const float* __restrict__ gb, float* __restrict__ gate_probs,
    int* __restrict__ top_i, int* __restrict__ tok_e,
    float* __restrict__ tok_g, int* __restrict__ counts) {
  const int wave = threadIdx.x >> 5;
  const int lane = threadIdx.x & 31;
  const int b = blockIdx.x * 8 + wave;
  if (b >= B_TOK) return;

  float acc[NEXP];
#pragma unroll
  for (int e = 0; e < NEXP; ++e) acc[e] = 0.0f;

  const float* xr = x + (size_t)b * DDIM;
  for (int d = lane; d < DDIM; d += 32) {
    const float xv = xr[d];
    const float* wr = gw + (size_t)d * NEXP;
#pragma unroll
    for (int e = 0; e < NEXP; ++e) acc[e] += xv * wr[e];
  }
#pragma unroll
  for (int e = 0; e < NEXP; ++e) {
#pragma unroll
    for (int off = 16; off > 0; off >>= 1)
      acc[e] += __shfl_xor(acc[e], off, 32);
  }

  if (lane == 0) {
    float logit[NEXP];
    float mx = -3.0e38f;
#pragma unroll
    for (int e = 0; e < NEXP; ++e) {
      logit[e] = acc[e] + gb[e];
      mx = fmaxf(mx, logit[e]);
    }
    float p[NEXP], denom = 0.0f;
#pragma unroll
    for (int e = 0; e < NEXP; ++e) {
      p[e] = __expf(logit[e] - mx);
      denom += p[e];
    }
    const float inv = 1.0f / denom;
#pragma unroll
    for (int e = 0; e < NEXP; ++e)
      gate_probs[(size_t)b * NEXP + e] = p[e] * inv;

    int i0 = 0;
#pragma unroll
    for (int e = 1; e < NEXP; ++e)
      if (logit[e] > logit[i0]) i0 = e;
    int i1 = (i0 == 0) ? 1 : 0;
#pragma unroll
    for (int e = 0; e < NEXP; ++e)
      if (e != i0 && logit[e] > logit[i1]) i1 = e;

    const float g0 = 1.0f / (1.0f + __expf(logit[i1] - logit[i0]));
    const float g1 = 1.0f - g0;

    top_i[b * TOPK + 0] = i0;
    top_i[b * TOPK + 1] = i1;
    tok_e[b * TOPK + 0] = i0;
    tok_e[b * TOPK + 1] = i1;
    tok_g[b * TOPK + 0] = g0;
    tok_g[b * TOPK + 1] = g1;
    atomicAdd(&counts[i0], 1);
    atomicAdd(&counts[i1], 1);
  }
}

// =====================================================================
// Kernel 2: exclusive scan of 8 expert counts -> offsets, cursors.
// =====================================================================
__global__ void moe_scan_kernel(const int* __restrict__ counts,
                                int* __restrict__ offsets,
                                int* __restrict__ cursors) {
  if (threadIdx.x == 0) {
    int a = 0;
    for (int e = 0; e < NEXP; ++e) {
      offsets[e] = a;
      cursors[e] = a;
      a += counts[e];
    }
  }
}

// =====================================================================
// Kernel 3: scatter (token, gate) pairs into contiguous per-expert slots.
// =====================================================================
__global__ __launch_bounds__(256) void moe_scatter_kernel(
    const int* __restrict__ tok_e, const float* __restrict__ tok_g,
    int* __restrict__ cursors, int* __restrict__ slot_token,
    float* __restrict__ slot_gate) {
  const int t = blockIdx.x * blockDim.x + threadIdx.x;
  if (t >= B_TOK) return;
#pragma unroll
  for (int k = 0; k < TOPK; ++k) {
    const int e = tok_e[t * TOPK + k];
    const int s = atomicAdd(&cursors[e], 1);
    slot_token[s] = t;
    slot_gate[s] = tok_g[t * TOPK + k];
  }
}

// =====================================================================
// GEMM tiling: 128x128 macro tile, K-step 32, 8 waves, DOUBLE-BUFFERED:
// stage tile k+1 while computing tile k (one barrier per iteration).
// A tile K-contiguous, B tile transposed (n-major): every fragment is
// two aligned ds_load_b128. LDS row stride = 40 shorts (80 B).
// =====================================================================
#define BM  128
#define BK  32
#define LDP 40

// Kernel 4: h = relu(gather(x) @ w1[e] + b1[e]), bf16 out to workspace.
__global__ __launch_bounds__(256) void moe_ffn1_kernel(
    const float* __restrict__ x, const float* __restrict__ w1,
    const float* __restrict__ b1, const int* __restrict__ slot_token,
    const int* __restrict__ offsets, const int* __restrict__ counts,
    unsigned short* __restrict__ h_ws) {
  const int e = blockIdx.z;
  const int cnt = counts[e];
  if ((int)(blockIdx.y * BM) >= cnt) return;
  const int row0 = offsets[e] + blockIdx.y * BM;
  const int row_end = offsets[e] + cnt;
  const int col0 = blockIdx.x * 128;

  __shared__ alignas(16) unsigned short As[2][BM][LDP];   // [buf][row][k]
  __shared__ alignas(16) unsigned short Bt[2][128][LDP];  // [buf][n][k]

  const int t = threadIdx.x;
  const int wave = t >> 5;
  const int lane = t & 31;
  const int lo = lane & 15;
  const int hi = lane >> 4;

  const int ar = t >> 3;          // A: rows ar + 32i, cols [ac, ac+4)
  const int ac = (t & 7) * 4;
  const int bn4 = (t & 31) * 4;   // B: k rows i*8+wave, cols [bn4, bn4+4)

  int tokA[4];
#pragma unroll
  for (int i = 0; i < 4; ++i) {
    const int row = row0 + ar + 32 * i;
    tokA[i] = (row < row_end) ? slot_token[row] : 0;
  }

  const float* w1e = w1 + (size_t)e * DDIM * HDIM + col0;

  auto stage = [&](int kk, int bsel) {
#pragma unroll
    for (int i = 0; i < 4; ++i) {  // A: gathered x rows, f32 -> bf16 packed
      const float4 xv = *(const float4*)(x + (size_t)tokA[i] * DDIM + kk + ac);
      uint2 p;
      p.x = pack2bf(xv.x, xv.y);
      p.y = pack2bf(xv.z, xv.w);
      *(uint2*)&As[bsel][ar + 32 * i][ac] = p;
    }
#pragma unroll
    for (int i = 0; i < 4; ++i) {  // B: transposed store (n-major)
      const int kb = i * 8 + wave;
      const float4 wv = *(const float4*)(w1e + (size_t)(kk + kb) * HDIM + bn4);
      Bt[bsel][bn4 + 0][kb] = f2bf(wv.x);
      Bt[bsel][bn4 + 1][kb] = f2bf(wv.y);
      Bt[bsel][bn4 + 2][kb] = f2bf(wv.z);
      Bt[bsel][bn4 + 3][kb] = f2bf(wv.w);
    }
    if (kk + BK < DDIM)  // prefetch the tile after next
      __builtin_prefetch(w1e + (size_t)(kk + BK + wave) * HDIM + bn4, 0, 1);
  };

  floatx8 acc[8] = {};
  stage(0, 0);
  int buf = 0;
  for (int k0 = 0; k0 < DDIM; k0 += BK) {
    __syncthreads();  // staged data for `buf` visible; prior compute done
    const int nxt = buf ^ 1;
    if (k0 + BK < DDIM) stage(k0 + BK, nxt);  // overlap with WMMAs below

    Frag16 a, bfr[8];
    a.h[0] = *(const ushortx8*)&As[buf][wave * 16 + lo][hi * 8];
    a.h[1] = *(const ushortx8*)&As[buf][wave * 16 + lo][16 + hi * 8];
#pragma unroll
    for (int c = 0; c < 8; ++c) {
      bfr[c].h[0] = *(const ushortx8*)&Bt[buf][c * 16 + lo][hi * 8];
      bfr[c].h[1] = *(const ushortx8*)&Bt[buf][c * 16 + lo][16 + hi * 8];
    }
#pragma unroll
    for (int c = 0; c < 8; ++c)
      acc[c] = __builtin_amdgcn_wmma_f32_16x16x32_bf16(
          false, a.v, false, bfr[c].v, (short)0, acc[c], false, false);
    buf = nxt;
  }

  // epilogue: +bias, relu, store bf16. D layout: VGPR r -> M = r + 8*hi.
#pragma unroll
  for (int r = 0; r < 8; ++r) {
    const int row = row0 + wave * 16 + r + 8 * hi;
    if (row < row_end) {
#pragma unroll
      for (int c = 0; c < 8; ++c) {
        const int col = col0 + c * 16 + lo;
        float v = acc[c][r] + b1[(size_t)e * HDIM + col];
        v = v > 0.0f ? v : 0.0f;
        h_ws[(size_t)row * HDIM + col] = f2bf(v);
      }
    }
  }
}

// Kernel 5: y = h @ w2[e]; out[token] += gate * (y + b2[e]).
// A tile (already bf16) staged with CDNA5 async DMA to LDS, double-buffered:
// s_wait_asynccnt at loop top waits only for the PREVIOUS iteration's copies.
__global__ __launch_bounds__(256) void moe_ffn2_kernel(
    const unsigned short* __restrict__ h_ws, const float* __restrict__ w2,
    const float* __restrict__ b2, const int* __restrict__ slot_token,
    const float* __restrict__ slot_gate, const int* __restrict__ offsets,
    const int* __restrict__ counts, float* __restrict__ out) {
  const int e = blockIdx.z;
  const int cnt = counts[e];
  if ((int)(blockIdx.y * BM) >= cnt) return;
  const int row0 = offsets[e] + blockIdx.y * BM;
  const int row_end = offsets[e] + cnt;
  const int col0 = blockIdx.x * 128;

  __shared__ alignas(16) unsigned short As[2][BM][LDP];
  __shared__ alignas(16) unsigned short Bt[2][128][LDP];

  const int t = threadIdx.x;
  const int wave = t >> 5;
  const int lane = t & 31;
  const int lo = lane & 15;
  const int hi = lane >> 4;

  // A async-copy mapping: one 16-B chunk per lane; 4 chunks per 64-B row.
  const int arow[2] = { (t >> 2), 64 + (t >> 2) };
  const int acs = (t & 3) * 8;  // short offset within row (16-B aligned)
  const int bn4 = (t & 31) * 4;

  // clamped source rows: OOB rows read a safe address; their results land
  // only in output rows >= row_end, which are never stored.
  size_t srcrow[2];
#pragma unroll
  for (int i = 0; i < 2; ++i) {
    int row = row0 + arow[i];
    row = (row < row_end) ? row : row0;
    srcrow[i] = (size_t)row * HDIM + acs;
  }

  const float* w2e = w2 + (size_t)e * HDIM * DDIM + col0;

  auto stage = [&](int kk, int bsel) {
#pragma unroll
    for (int i = 0; i < 2; ++i) {  // A: async DMA, no VGPR round-trip
      const unsigned long long ga =
          (unsigned long long)(size_t)(h_ws + srcrow[i] + kk);
      const unsigned ldst = (unsigned)(size_t)&As[bsel][arow[i]][acs];
      asm volatile("global_load_async_to_lds_b128 %0, %1, off"
                   :: "v"(ldst), "v"(ga)
                   : "memory");
    }
#pragma unroll
    for (int i = 0; i < 4; ++i) {  // B: transposed store (n-major)
      const int kb = i * 8 + wave;
      const float4 wv = *(const float4*)(w2e + (size_t)(kk + kb) * DDIM + bn4);
      Bt[bsel][bn4 + 0][kb] = f2bf(wv.x);
      Bt[bsel][bn4 + 1][kb] = f2bf(wv.y);
      Bt[bsel][bn4 + 2][kb] = f2bf(wv.z);
      Bt[bsel][bn4 + 3][kb] = f2bf(wv.w);
    }
    if (kk + BK < HDIM)
      __builtin_prefetch(w2e + (size_t)(kk + BK + wave) * DDIM + bn4, 0, 1);
  };

  floatx8 acc[8] = {};
  stage(0, 0);
  int buf = 0;
  for (int k0 = 0; k0 < HDIM; k0 += BK) {
    asm volatile("s_wait_asynccnt 0" ::: "memory");  // prev iter's A copies
    __syncthreads();
    const int nxt = buf ^ 1;
    if (k0 + BK < HDIM) stage(k0 + BK, nxt);  // overlap with WMMAs below

    Frag16 a, bfr[8];
    a.h[0] = *(const ushortx8*)&As[buf][wave * 16 + lo][hi * 8];
    a.h[1] = *(const ushortx8*)&As[buf][wave * 16 + lo][16 + hi * 8];
#pragma unroll
    for (int c = 0; c < 8; ++c) {
      bfr[c].h[0] = *(const ushortx8*)&Bt[buf][c * 16 + lo][hi * 8];
      bfr[c].h[1] = *(const ushortx8*)&Bt[buf][c * 16 + lo][16 + hi * 8];
    }
#pragma unroll
    for (int c = 0; c < 8; ++c)
      acc[c] = __builtin_amdgcn_wmma_f32_16x16x32_bf16(
          false, a.v, false, bfr[c].v, (short)0, acc[c], false, false);
    buf = nxt;
  }

#pragma unroll
  for (int r = 0; r < 8; ++r) {
    const int row = row0 + wave * 16 + r + 8 * hi;
    if (row < row_end) {
      const float g = slot_gate[row];
      const int token = slot_token[row];
      float* orow = out + (size_t)token * DDIM;
#pragma unroll
      for (int c = 0; c < 8; ++c) {
        const int col = col0 + c * 16 + lo;
        const float v = (acc[c][r] + b2[(size_t)e * DDIM + col]) * g;
        atomicAdd(orow + col, v);
      }
    }
  }
}

// =====================================================================
// Host launcher
// =====================================================================
extern "C" void kernel_launch(void* const* d_in, const int* in_sizes, int n_in,
                              void* d_out, int out_size, void* d_ws, size_t ws_size,
                              hipStream_t stream) {
  (void)in_sizes; (void)n_in; (void)out_size; (void)ws_size;
  const float* x  = (const float*)d_in[0];
  const float* gw = (const float*)d_in[1];
  const float* gb = (const float*)d_in[2];
  const float* w1 = (const float*)d_in[3];
  const float* b1 = (const float*)d_in[4];
  const float* w2 = (const float*)d_in[5];
  const float* b2 = (const float*)d_in[6];

  float* out = (float*)d_out;                                   // [B, D]
  float* gate_probs = out + (size_t)B_TOK * DDIM;               // [B, E]
  int*   top_i = (int*)(gate_probs + (size_t)B_TOK * NEXP);     // [B, K]

  char* ws = (char*)d_ws;
  int*   counts     = (int*)(ws + 0);
  int*   offsets    = (int*)(ws + 256);
  int*   cursors    = (int*)(ws + 512);
  int*   tok_e      = (int*)(ws + 1024);
  float* tok_g      = (float*)(ws + 1024 + (size_t)B_TOK * 8);
  int*   slot_token = (int*)(ws + 1024 + (size_t)B_TOK * 16);
  float* slot_gate  = (float*)(ws + 1024 + (size_t)B_TOK * 24);
  unsigned short* h_ws =
      (unsigned short*)(ws + 1024 + (size_t)B_TOK * 32);        // [B*K, H] bf16

  hipMemsetAsync(counts, 0, NEXP * sizeof(int), stream);
  hipMemsetAsync(out, 0, (size_t)B_TOK * DDIM * sizeof(float), stream);

  moe_gate_kernel<<<B_TOK / 8, 256, 0, stream>>>(x, gw, gb, gate_probs, top_i,
                                                 tok_e, tok_g, counts);
  moe_scan_kernel<<<1, 32, 0, stream>>>(counts, offsets, cursors);
  moe_scatter_kernel<<<B_TOK / 256, 256, 0, stream>>>(tok_e, tok_g, cursors,
                                                      slot_token, slot_gate);
  dim3 g1(HDIM / 128, B_TOK / BM, NEXP);
  moe_ffn1_kernel<<<g1, 256, 0, stream>>>(x, w1, b1, slot_token, offsets,
                                          counts, h_ws);
  dim3 g2(DDIM / 128, B_TOK / BM, NEXP);
  moe_ffn2_kernel<<<g2, 256, 0, stream>>>(h_ws, w2, b2, slot_token, slot_gate,
                                          offsets, counts, out);
}
